// ARCformer_33732673143385
// MI455X (gfx1250) — matrix operand order repaired and checked by reading, not compiled, not run on hardware
//
#include <hip/hip_runtime.h>

// ---------------------------------------------------------------------------
// ARCformer block for MI455X (gfx1250, wave32, WMMA bf16 + async-LDS path)
// ---------------------------------------------------------------------------
#define BB 4
#define LL 1024
#define SS 256
#define DD 2048
#define HH 16
#define DKV 128
#define TT (SS + LL + SS)   /* 1536 */
#define DHH 8192

typedef unsigned short u16;
typedef unsigned int   u32;
typedef __attribute__((ext_vector_type(16))) __bf16 v16bf;
typedef __attribute__((ext_vector_type(8)))  float  v8f;

union Frag16 {
  v16bf v;
  uint4 q[2];
  u16   u[16];
};

__device__ __forceinline__ u16 f32_to_bf16(float f) {
  u32 x = __float_as_uint(f);
  u32 r = x + 0x7FFFu + ((x >> 16) & 1u);   // round-to-nearest-even
  return (u16)(r >> 16);
}

// Async global -> LDS copy, 16 bytes per lane, tracked by ASYNCcnt.
// GV addressing mode: 64-bit address in VGPR pair, LDS byte offset in VDST.
__device__ __forceinline__ void async_copy_b128(void* lds, const void* gptr) {
  u32 l = (u32)(unsigned long long)lds;     // low 32 bits = LDS offset
  asm volatile("global_load_async_to_lds_b128 %0, %1, off"
               :: "v"(l), "v"(gptr) : "memory");
}
__device__ __forceinline__ void wait_async0() {
  asm volatile("s_wait_asynccnt 0x0" ::: "memory");
}

// ---------------------------------------------------------------------------
// Fused fp32 -> bf16 cast + transpose (src: K x N, dst: N x K), LDS-tiled
// ---------------------------------------------------------------------------
__global__ __launch_bounds__(256) void cast_transpose_kernel(
    const float* __restrict__ src, u16* __restrict__ dst, int K, int N) {
  __shared__ float tile[32][33];
  const int k0 = blockIdx.y * 32, n0 = blockIdx.x * 32;
  const int tx = threadIdx.x, ty = threadIdx.y;   // (32, 8)
#pragma unroll
  for (int i = ty; i < 32; i += 8)
    tile[i][tx] = src[(size_t)(k0 + i) * N + n0 + tx];
  __syncthreads();
#pragma unroll
  for (int i = ty; i < 32; i += 8)
    dst[(size_t)(n0 + i) * K + k0 + tx] = f32_to_bf16(tile[tx][i]);
}

// seq = concat([state, x, state], axis=1), cast to bf16
__global__ void build_seq_kernel(const float* __restrict__ x,
                                 const float* __restrict__ st,
                                 u16* __restrict__ seq) {
  const long long n = (long long)BB * TT * DD;
  long long i = (long long)blockIdx.x * blockDim.x + threadIdx.x;
  long long stride = (long long)gridDim.x * blockDim.x;
  for (; i < n; i += stride) {
    int d = (int)(i % DD);
    long long row = i / DD;
    int t = (int)(row % TT);
    int b = (int)(row / TT);
    float v;
    if (t < SS)            v = st[((long long)b * SS + t) * DD + d];
    else if (t < SS + LL)  v = x [((long long)b * LL + (t - SS)) * DD + d];
    else                   v = st[((long long)b * SS + (t - SS - LL)) * DD + d];
    seq[i] = f32_to_bf16(v);
  }
}

// ---------------------------------------------------------------------------
// WMMA GEMM:  C[M,N] = A[M,K](bf16) @ B[K,N](bf16) + bias
// B is passed PRE-TRANSPOSED (Bt: N x K row-major) so both tiles are
// contiguity-preserving async global->LDS B128 copies.
// Block tile 256x128, BK=32, 256 threads = 8 waves (4m x 2n), wave tile
// 64x64 -> 16 WMMAs per wave per k-step. Double-buffered LDS with
// ASYNCcnt-tracked prefetch of tile t+1 overlapping WMMA on tile t.
// ---------------------------------------------------------------------------
template <bool GELU_ACT, bool OUT_BF16>
__global__ __launch_bounds__(256) void gemm_kernel(
    const u16* __restrict__ A, const u16* __restrict__ Bt,
    const float* __restrict__ bias, void* __restrict__ Cout,
    int M, int N, int K) {
  __shared__ __align__(16) u16 As[2][256][40];   // [m][k], 80B row stride
  __shared__ __align__(16) u16 Bs[2][128][40];   // [n][k]

  const int t    = threadIdx.x;
  const int wave = t >> 5, lane = t & 31;
  const int g    = lane >> 4, ln = lane & 15;
  const int wm   = wave >> 1, wn = wave & 1;     // 4 x 2 wave grid
  const int m0   = blockIdx.y * 256;
  const int n0   = blockIdx.x * 128;

  // issue async staging of one BK=32 tile pair into buffer `buf`
  auto stage = [&](int buf, int k0) {
#pragma unroll
    for (int it = 0; it < 4; ++it) {             // A: 256x32 = 1024 x B128
      int idx = t + it * 256;
      int r = idx >> 2, c = (idx & 3) << 3;
      async_copy_b128(&As[buf][r][c], A + (size_t)(m0 + r) * K + k0 + c);
    }
#pragma unroll
    for (int it = 0; it < 2; ++it) {             // B: 128x32 = 512 x B128
      int idx = t + it * 256;
      int r = idx >> 2, c = (idx & 3) << 3;
      async_copy_b128(&Bs[buf][r][c], Bt + (size_t)(n0 + r) * K + k0 + c);
    }
  };

  v8f acc[4][4];
#pragma unroll
  for (int i = 0; i < 4; ++i)
#pragma unroll
    for (int j = 0; j < 4; ++j) acc[i][j] = (v8f){0,0,0,0,0,0,0,0};

  stage(0, 0);
  wait_async0();
  __syncthreads();

  int buf = 0;
  for (int k0 = 0; k0 < K; k0 += 32) {
    if (k0 + 32 < K) stage(buf ^ 1, k0 + 32);    // prefetch overlaps WMMA

    Frag16 bf[4];
#pragma unroll
    for (int nt = 0; nt < 4; ++nt) {
      const u16* p = &Bs[buf][wn * 64 + nt * 16 + ln][g * 8];
      bf[nt].q[0] = *(const uint4*)p;
      bf[nt].q[1] = *(const uint4*)(p + 16);
    }
#pragma unroll
    for (int mt = 0; mt < 4; ++mt) {
      Frag16 af;
      const u16* p = &As[buf][wm * 64 + mt * 16 + ln][0];
      af.q[0] = *(const uint4*)(p + g * 8);
      af.q[1] = *(const uint4*)(p + 16 + g * 8);
#pragma unroll
      for (int nt = 0; nt < 4; ++nt) {
        acc[mt][nt] = __builtin_amdgcn_wmma_f32_16x16x32_bf16(
            false, af.v, false, bf[nt].v, (short)0, acc[mt][nt], false, false);
      }
    }

    wait_async0();      // own prefetch landed; barrier covers other waves
    __syncthreads();
    buf ^= 1;
  }

  // epilogue: bias (+GELU), f32 or bf16 store; C layout: VGPR r -> M=r+8g
#pragma unroll
  for (int mt = 0; mt < 4; ++mt) {
#pragma unroll
    for (int nt = 0; nt < 4; ++nt) {
      int col  = n0 + wn * 64 + nt * 16 + ln;
      float bv = bias ? bias[col] : 0.0f;
#pragma unroll
      for (int r = 0; r < 8; ++r) {
        int row = m0 + wm * 64 + mt * 16 + r + 8 * g;
        float v = acc[mt][nt][r] + bv;
        if (GELU_ACT) {
          float x3 = v * v * v;
          v = 0.5f * v * (1.0f + tanhf(0.7978845608f * (v + 0.044715f * x3)));
        }
        if (OUT_BF16)
          ((u16*)Cout)[(size_t)row * N + col] = f32_to_bf16(v);
        else
          ((float*)Cout)[(size_t)row * N + col] = v;
      }
    }
  }
}

// ---------------------------------------------------------------------------
// Flash attention (causal over T=1536), one block per (b, h, 128 q rows),
// 8 waves each owning a 16-row q-tile. K/V 32x128 tiles staged in LDS.
// S = Q K^T via v_wmma, online softmax, P via per-wave LDS round-trip
// (C-layout -> A-layout), O += P V via v_wmma.
// ---------------------------------------------------------------------------
__global__ __launch_bounds__(256) void attn_kernel(
    const u16* __restrict__ qb, const u16* __restrict__ kb,
    const u16* __restrict__ vb, u16* __restrict__ ob) {
  __shared__ __align__(16) u16 Ks[32][136];     // [key][d]
  __shared__ __align__(16) u16 Vs[128][40];     // [d][key] (transposed)
  __shared__ __align__(16) u16 Ps[8][16][40];   // per-wave P scratch

  const int t    = threadIdx.x;
  const int wave = t >> 5, lane = t & 31;
  const int g    = lane >> 4, ln = lane & 15;
  const int b = blockIdx.z, h = blockIdx.y, qblk = blockIdx.x;
  const int qt0 = qblk * 128 + wave * 16;
  const float scale = 0.08838834764831845f;     // 1/sqrt(128)

  Frag16 qf[4];
  {
    const u16* qrow = qb + ((size_t)b * TT + qt0 + ln) * DD + h * DKV;
#pragma unroll
    for (int c = 0; c < 4; ++c) {
      qf[c].q[0] = *(const uint4*)(qrow + c * 32 + g * 8);
      qf[c].q[1] = *(const uint4*)(qrow + c * 32 + 16 + g * 8);
    }
  }

  float mrow[8], lrow[8];
  v8f oacc[8];
#pragma unroll
  for (int r = 0; r < 8; ++r) { mrow[r] = -1e30f; lrow[r] = 0.0f; }
#pragma unroll
  for (int j = 0; j < 8; ++j) oacc[j] = (v8f){0,0,0,0,0,0,0,0};

  const int kend = qblk * 128 + 128;
  for (int kt0 = 0; kt0 < kend; kt0 += 32) {
    __syncthreads();
    {
      int r = t >> 3;                 // K tile 32x128
      int c = (t & 7) << 4;
      const u16* src = kb + ((size_t)b * TT + kt0 + r) * DD + h * DKV + c;
      *(uint4*)&Ks[r][c]     = *(const uint4*)src;
      *(uint4*)&Ks[r][c + 8] = *(const uint4*)(src + 8);
    }
#pragma unroll
    for (int it = 0; it < 2; ++it) {  // V tile 32x128 transposed
      int idx = t + it * 256;
      int kk  = idx >> 4;
      int nn  = (idx & 15) << 3;
      union { uint4 q; u16 u[8]; } tmp;
      tmp.q = *(const uint4*)(vb + ((size_t)b * TT + kt0 + kk) * DD +
                              h * DKV + nn);
#pragma unroll
      for (int i = 0; i < 8; ++i) Vs[nn + i][kk] = tmp.u[i];
    }
    __syncthreads();

    if (kt0 <= qt0 + 15) {
      v8f s0 = (v8f){0,0,0,0,0,0,0,0};
      v8f s1 = (v8f){0,0,0,0,0,0,0,0};
#pragma unroll
      for (int c = 0; c < 4; ++c) {
        Frag16 kf0, kf1;
        const u16* p0 = &Ks[ln][c * 32 + g * 8];
        const u16* p1 = &Ks[16 + ln][c * 32 + g * 8];
        kf0.q[0] = *(const uint4*)p0; kf0.q[1] = *(const uint4*)(p0 + 16);
        kf1.q[0] = *(const uint4*)p1; kf1.q[1] = *(const uint4*)(p1 + 16);
        s0 = __builtin_amdgcn_wmma_f32_16x16x32_bf16(
            false, qf[c].v, false, kf0.v, (short)0, s0, false, false);
        s1 = __builtin_amdgcn_wmma_f32_16x16x32_bf16(
            false, qf[c].v, false, kf1.v, (short)0, s1, false, false);
      }
#pragma unroll
      for (int r = 0; r < 8; ++r) {
        int qg = qt0 + r + 8 * g;
        float a0 = s0[r] * scale;
        float a1 = s1[r] * scale;
        if (kt0 + ln > qg)      a0 = -1e30f;
        if (kt0 + 16 + ln > qg) a1 = -1e30f;
        float mx = fmaxf(a0, a1);
#pragma unroll
        for (int off = 1; off < 16; off <<= 1)
          mx = fmaxf(mx, __shfl_xor(mx, off, 32));
        float mnew = fmaxf(mrow[r], mx);
        float p0 = __expf(a0 - mnew);
        float p1 = __expf(a1 - mnew);
        float rs = p0 + p1;
#pragma unroll
        for (int off = 1; off < 16; off <<= 1)
          rs += __shfl_xor(rs, off, 32);
        float corr = __expf(mrow[r] - mnew);
        lrow[r] = lrow[r] * corr + rs;
        mrow[r] = mnew;
#pragma unroll
        for (int j = 0; j < 8; ++j) oacc[j][r] *= corr;
        Ps[wave][r + 8 * g][ln]      = f32_to_bf16(p0);
        Ps[wave][r + 8 * g][16 + ln] = f32_to_bf16(p1);
      }
      asm volatile("s_wait_dscnt 0x0" ::: "memory");  // P store->load, same wave
      Frag16 pf;
      const u16* pp = &Ps[wave][ln][0];
      pf.q[0] = *(const uint4*)(pp + g * 8);
      pf.q[1] = *(const uint4*)(pp + 16 + g * 8);
#pragma unroll
      for (int j = 0; j < 8; ++j) {
        Frag16 vf;
        const u16* vp = &Vs[j * 16 + ln][g * 8];
        vf.q[0] = *(const uint4*)vp;
        vf.q[1] = *(const uint4*)(vp + 16);
        oacc[j] = __builtin_amdgcn_wmma_f32_16x16x32_bf16(
            false, pf.v, false, vf.v, (short)0, oacc[j], false, false);
      }
    }
  }

#pragma unroll
  for (int r = 0; r < 8; ++r) {
    float inv = 1.0f / lrow[r];
    int row   = qt0 + r + 8 * g;
#pragma unroll
    for (int j = 0; j < 8; ++j) {
      ob[((size_t)b * TT + row) * DD + h * DKV + j * 16 + ln] =
          f32_to_bf16(oacc[j][r] * inv);
    }
  }
}

// ---------------------------------------------------------------------------
// h = LN(attn_out + x); emits f32 (residual path) and bf16 (GEMM input)
// ---------------------------------------------------------------------------
__global__ __launch_bounds__(256) void add_ln_kernel(
    const float* __restrict__ full, const float* __restrict__ x,
    const float* __restrict__ gam, const float* __restrict__ bet,
    float* __restrict__ hf, u16* __restrict__ hb) {
  __shared__ float red[16];
  const int row = blockIdx.x;
  const int b = row / LL, l = row % LL;
  const float* a  = full + ((size_t)b * TT + SS + l) * DD;
  const float* xr = x + (size_t)row * DD;
  const int t = threadIdx.x;
  float sum = 0.0f, sq = 0.0f;
  for (int i = t; i < DD; i += 256) {
    float v = a[i] + xr[i];
    sum += v; sq += v * v;
  }
#pragma unroll
  for (int off = 1; off < 32; off <<= 1) {
    sum += __shfl_xor(sum, off, 32);
    sq  += __shfl_xor(sq, off, 32);
  }
  const int wave = t >> 5, lane = t & 31;
  if (lane == 0) { red[wave] = sum; red[8 + wave] = sq; }
  __syncthreads();
  if (t == 0) {
    float s = 0, q = 0;
    for (int w = 0; w < 8; ++w) { s += red[w]; q += red[8 + w]; }
    red[0] = s; red[8] = q;
  }
  __syncthreads();
  float mean = red[0] / (float)DD;
  float var  = red[8] / (float)DD - mean * mean;
  float inv  = rsqrtf(var + 1e-5f);
  for (int i = t; i < DD; i += 256) {
    float v = a[i] + xr[i];
    float o = (v - mean) * inv * gam[i] + bet[i];
    hf[(size_t)row * DD + i] = o;
    hb[(size_t)row * DD + i] = f32_to_bf16(o);
  }
}

// out = LN(mlp + h) -> d_out (f32)
__global__ __launch_bounds__(256) void final_ln_kernel(
    const float* __restrict__ mlp, const float* __restrict__ hf,
    const float* __restrict__ gam, const float* __restrict__ bet,
    float* __restrict__ out) {
  __shared__ float red[16];
  const int row = blockIdx.x;
  const float* a  = mlp + (size_t)row * DD;
  const float* xr = hf + (size_t)row * DD;
  const int t = threadIdx.x;
  float sum = 0.0f, sq = 0.0f;
  for (int i = t; i < DD; i += 256) {
    float v = a[i] + xr[i];
    sum += v; sq += v * v;
  }
#pragma unroll
  for (int off = 1; off < 32; off <<= 1) {
    sum += __shfl_xor(sum, off, 32);
    sq  += __shfl_xor(sq, off, 32);
  }
  const int wave = t >> 5, lane = t & 31;
  if (lane == 0) { red[wave] = sum; red[8 + wave] = sq; }
  __syncthreads();
  if (t == 0) {
    float s = 0, q = 0;
    for (int w = 0; w < 8; ++w) { s += red[w]; q += red[8 + w]; }
    red[0] = s; red[8] = q;
  }
  __syncthreads();
  float mean = red[0] / (float)DD;
  float var  = red[8] / (float)DD - mean * mean;
  float inv  = rsqrtf(var + 1e-5f);
  for (int i = t; i < DD; i += 256) {
    float v = a[i] + xr[i];
    out[(size_t)row * DD + i] = (v - mean) * inv * gam[i] + bet[i];
  }
}

// state_next = full[:, S+L:, :]
__global__ void copy_state_kernel(const float* __restrict__ full,
                                  float* __restrict__ dst) {
  const long long n = (long long)BB * SS * DD;
  long long i = (long long)blockIdx.x * blockDim.x + threadIdx.x;
  long long stride = (long long)gridDim.x * blockDim.x;
  for (; i < n; i += stride) {
    int d = (int)(i % DD);
    long long r = i / DD;
    int s = (int)(r % SS);
    int b = (int)(r / SS);
    dst[i] = full[((size_t)b * TT + SS + LL + s) * DD + d];
  }
}

// ---------------------------------------------------------------------------
extern "C" void kernel_launch(void* const* d_in, const int* in_sizes, int n_in,
                              void* d_out, int out_size, void* d_ws,
                              size_t ws_size, hipStream_t stream) {
  (void)in_sizes; (void)out_size; (void)ws_size;
  if (n_in < 20 || !d_ws) return;

  const float* x   = (const float*)d_in[0];
  const float* st  = (const float*)d_in[1];
  const float* Wq  = (const float*)d_in[2];
  const float* bq  = (const float*)d_in[3];
  const float* Wk  = (const float*)d_in[4];
  const float* bk  = (const float*)d_in[5];
  const float* Wv  = (const float*)d_in[6];
  const float* bv  = (const float*)d_in[7];
  const float* Wo1 = (const float*)d_in[8];
  const float* bo1 = (const float*)d_in[9];
  const float* Wo2 = (const float*)d_in[10];
  const float* bo2 = (const float*)d_in[11];
  const float* g1  = (const float*)d_in[12];
  const float* b1  = (const float*)d_in[13];
  const float* Wm1 = (const float*)d_in[14];
  const float* bm1 = (const float*)d_in[15];
  const float* Wm2 = (const float*)d_in[16];
  const float* bm2 = (const float*)d_in[17];
  const float* g2  = (const float*)d_in[18];
  const float* b2  = (const float*)d_in[19];

  const long long BT = (long long)BB * TT;   // 6144
  const long long BL = (long long)BB * LL;   // 4096

  char* ws = (char*)d_ws;
  size_t off = 0;
  auto alloc = [&](size_t bytes) -> void* {
    void* p = ws + off;
    off = (off + bytes + 255) & ~(size_t)255;
    return p;
  };

  u16* seqb  = (u16*)alloc((size_t)BT * DD * 2);
  u16* wqb   = (u16*)alloc((size_t)DD * DD * 2);    // all weights stored N x K
  u16* wkb   = (u16*)alloc((size_t)DD * DD * 2);
  u16* wvb   = (u16*)alloc((size_t)DD * DD * 2);
  u16* wo1b  = (u16*)alloc((size_t)DD * DD * 2);
  u16* wo2b  = (u16*)alloc((size_t)DD * DD * 2);
  u16* wm1b  = (u16*)alloc((size_t)DD * DHH * 2);
  u16* wm2b  = (u16*)alloc((size_t)DHH * DD * 2);
  u16* qbuf  = (u16*)alloc((size_t)BT * DD * 2);
  u16* kbuf  = (u16*)alloc((size_t)BT * DD * 2);
  u16* vbuf  = (u16*)alloc((size_t)BT * DD * 2);
  u16* obuf  = (u16*)alloc((size_t)BT * DD * 2);
  u16* t1    = (u16*)alloc((size_t)BT * DD * 2);
  float* fullf = (float*)alloc((size_t)BT * DD * 4);
  float* hf    = (float*)alloc((size_t)BL * DD * 4);
  u16*   hb    = (u16*)alloc((size_t)BL * DD * 2);
  u16*   m1    = (u16*)alloc((size_t)BL * DHH * 2);
  float* m2    = (float*)alloc((size_t)BL * DD * 4);

  // 1) weight cast+transpose (K x N -> N x K, bf16) + seq build
  dim3 tb(32, 8);
  cast_transpose_kernel<<<dim3(DD / 32, DD / 32), tb, 0, stream>>>(Wq, wqb, DD, DD);
  cast_transpose_kernel<<<dim3(DD / 32, DD / 32), tb, 0, stream>>>(Wk, wkb, DD, DD);
  cast_transpose_kernel<<<dim3(DD / 32, DD / 32), tb, 0, stream>>>(Wv, wvb, DD, DD);
  cast_transpose_kernel<<<dim3(DD / 32, DD / 32), tb, 0, stream>>>(Wo1, wo1b, DD, DD);
  cast_transpose_kernel<<<dim3(DD / 32, DD / 32), tb, 0, stream>>>(Wo2, wo2b, DD, DD);
  cast_transpose_kernel<<<dim3(DHH / 32, DD / 32), tb, 0, stream>>>(Wm1, wm1b, DD, DHH);
  cast_transpose_kernel<<<dim3(DD / 32, DHH / 32), tb, 0, stream>>>(Wm2, wm2b, DHH, DD);
  build_seq_kernel<<<2048, 256, 0, stream>>>(x, st, seqb);

  // 2) QKV projections (M=6144, N=2048, K=2048), bf16 out
  dim3 blk(256);
  dim3 gqkv(DD / 128, (unsigned)(BT / 256));
  gemm_kernel<false, true><<<gqkv, blk, 0, stream>>>(seqb, wqb, bq, qbuf,
                                                     (int)BT, DD, DD);
  gemm_kernel<false, true><<<gqkv, blk, 0, stream>>>(seqb, wkb, bk, kbuf,
                                                     (int)BT, DD, DD);
  gemm_kernel<false, true><<<gqkv, blk, 0, stream>>>(seqb, wvb, bv, vbuf,
                                                     (int)BT, DD, DD);

  // 3) causal flash attention
  attn_kernel<<<dim3(TT / 128, HH, BB), blk, 0, stream>>>(qbuf, kbuf, vbuf,
                                                          obuf);

  // 4) output projections: t1 = o@Wo1+bo1 (bf16), full = t1@Wo2+bo2 (f32)
  gemm_kernel<false, true><<<gqkv, blk, 0, stream>>>(obuf, wo1b, bo1, t1,
                                                     (int)BT, DD, DD);
  gemm_kernel<false, false><<<gqkv, blk, 0, stream>>>(t1, wo2b, bo2, fullf,
                                                      (int)BT, DD, DD);

  // 5) state_next slice -> second half of d_out
  float* outp = (float*)d_out;
  copy_state_kernel<<<1024, 256, 0, stream>>>(fullf, outp + (size_t)BL * DD);

  // 6) h = LN(attn_out + x)
  add_ln_kernel<<<(unsigned)BL, blk, 0, stream>>>(fullf, x, g1, b1, hf, hb);

  // 7) MLP: m1 = gelu(h@Wm1+bm1) (bf16), m2 = m1@Wm2+bm2 (f32)
  gemm_kernel<true, true><<<dim3(DHH / 128, (unsigned)(BL / 256)), blk, 0,
                            stream>>>(hb, wm1b, bm1, m1, (int)BL, DHH, DD);
  gemm_kernel<false, false><<<dim3(DD / 128, (unsigned)(BL / 256)), blk, 0,
                              stream>>>(m1, wm2b, bm2, m2, (int)BL, DD, DHH);

  // 8) out = LN(m2 + h) -> first half of d_out
  final_ln_kernel<<<(unsigned)BL, blk, 0, stream>>>(m2, hf, g2, b2, outp);
}
